// LSTM_21577915695755
// MI455X (gfx1250) — matrix operand order repaired and checked by reading, not compiled, not run on hardware
//
#include <hip/hip_runtime.h>

// ---------------------------------------------------------------------------
// LSTM (B=64, T=2048, IN=768, H=100) + MLP head, for MI455X (gfx1250, wave32)
//   k_prep  : w_ih f32 -> f16 once (L2-resident, no inner-loop cvt)
//   k_xproj : 131072x768 @ 768x400 via v_wmma_f32_16x16x32_f16.
//             seq panel staged via TDM tensor_load_to_lds (TENSORcnt),
//             converted once to an f16 LDS panel; inner loop =
//             2 ds_load_b128 + 2 global_load_b128 + 1 wmma.
//   k_scan  : 4 workgroups x 16 batch rows; w_hh WMMA fragments + state in
//             LDS; per-step gate slice via GLOBAL_LOAD_ASYNC_TO_LDS_B128.
//   k_head  : relu(h @ w1^T + b1) @ w2^T + b2 -> out[64]
// ---------------------------------------------------------------------------

typedef __attribute__((ext_vector_type(16))) _Float16 v16h;
typedef __attribute__((ext_vector_type(8)))  float    v8f;
typedef __attribute__((ext_vector_type(4)))  unsigned u32x4;
typedef __attribute__((ext_vector_type(8)))  unsigned u32x8;

#define T_LEN  2048
#define IN_DIM 768
#define HDIM   100
#define G4     400      // 4*H
#define BATCH  64

union AF {
    v16h  h;
    uint4 q[2];
};

__device__ __forceinline__ float sigm(float x) {
    return 1.0f / (1.0f + __expf(-x));
}
__device__ __forceinline__ float tanh_fast(float x) {
    x = fminf(fmaxf(x, -15.0f), 15.0f);
    float e = __expf(2.0f * x);
    return (e - 1.0f) / (e + 1.0f);
}
__device__ __forceinline__ unsigned pack2h(float lo, float hi) {
    _Float16 l = (_Float16)lo, h = (_Float16)hi;
    unsigned short ul, uh;
    __builtin_memcpy(&ul, &l, 2);
    __builtin_memcpy(&uh, &h, 2);
    return (unsigned)ul | ((unsigned)uh << 16);
}

// --- CDNA5 async global->LDS copy (ASYNCcnt-tracked, §15.18 op 98) ---------
__device__ __forceinline__ void async_b128(unsigned lds_addr, const void* gaddr) {
    asm volatile("global_load_async_to_lds_b128 %0, %1, off"
                 :: "v"(lds_addr), "v"(gaddr) : "memory");
}
__device__ __forceinline__ void wait_async0() {
    asm volatile("s_wait_asynccnt 0x0" ::: "memory");
}
__device__ __forceinline__ unsigned lds_off32(const void* p) {
    return (unsigned)(size_t)p;   // low 32 bits of flat LDS address = LDS offset
}

// --- CDNA5 Tensor Data Mover: 2D f32 tile, global -> LDS (TENSORcnt) -------
// D# per ISA §8: group0 {count=1, lds_addr, global_addr[56:0], type=2},
// group1 {data_size=4B, tensor_dim0/tile_dim0=row_elems, tensor_dim1/
// tile_dim1=rows, tensor_dim0_stride=row_elems}; groups 2/3 zero (<=2D).
__device__ __forceinline__ void tdm_load_2d_f32(
    unsigned lds_addr, const void* gaddr, unsigned rows, unsigned row_elems)
{
    const unsigned long long ga = (unsigned long long)gaddr;
    u32x4 g0;
    g0[0] = 1u;                                   // count=1, user descriptor
    g0[1] = lds_addr;                             // LDS byte address
    g0[2] = (unsigned)ga;                         // global_addr[31:0]
    g0[3] = ((unsigned)(ga >> 32) & 0x1FFFFFFu)   // global_addr[56:32]
            | (2u << 30);                         // type = 2 ("image")
    u32x8 g1;
    g1[0] = 2u << 16;                             // data_size = 4 bytes
    g1[1] = (row_elems & 0xFFFFu) << 16;          // tensor_dim0[15:0]
    g1[2] = (row_elems >> 16) | (rows << 16);     // dim0[31:16] | dim1[15:0]
    g1[3] = (rows >> 16) | (row_elems << 16);     // dim1[31:16] | tile_dim0
    g1[4] = rows & 0xFFFFu;                       // tile_dim1 (tile_dim2 = 0)
    g1[5] = row_elems;                            // tensor_dim0_stride[31:0]
    g1[6] = 0u;
    g1[7] = 0u;
    u32x4 gz;
    gz[0] = 0u; gz[1] = 0u; gz[2] = 0u; gz[3] = 0u;
    asm volatile("tensor_load_to_lds %0, %1, %2, %3"
                 :: "s"(g0), "s"(g1), "s"(gz), "s"(gz) : "memory");
}

// ---------------------------------------------------------------------------
// Kernel P: w_ih f32 -> packed f16 pairs in workspace.
// ---------------------------------------------------------------------------
__global__ __launch_bounds__(256) void k_prep(
    const float* __restrict__ w_ih, unsigned* __restrict__ w16)
{
    const int idx = blockIdx.x * 256 + threadIdx.x;
    if (idx < G4 * IN_DIM / 2)
        w16[idx] = pack2h(w_ih[2 * idx], w_ih[2 * idx + 1]);
}

// ---------------------------------------------------------------------------
// Kernel A: big GEMM. Block = 256 threads (8 waves), 32 output rows.
// ---------------------------------------------------------------------------
__global__ __launch_bounds__(256) void k_xproj(
    const float* __restrict__ seq, const unsigned* __restrict__ w16,
    const float* __restrict__ b_ih, const float* __restrict__ b_hh,
    float* __restrict__ xp)
{
    __shared__ float    sAf[32 * IN_DIM];   // 96 KB raw f32 panel (TDM dst)
    __shared__ unsigned sAh[32 * 384];      // 48 KB f16 panel (WMMA source)
    const int  tid  = threadIdx.x;
    const long row0 = (long)blockIdx.x * 32;

    // Stage 32x768 f32 panel via one Tensor-Data-Mover descriptor (wave 0).
    if (tid < 32) {
        tdm_load_2d_f32(lds_off32(sAf), seq + row0 * IN_DIM, 32, IN_DIM);
        __builtin_amdgcn_s_wait_tensorcnt(0);
    }
    __syncthreads();
    // Convert once to f16 panel.
    for (int u = tid; u < 32 * 384; u += 256)
        sAh[u] = pack2h(sAf[2 * u], sAf[2 * u + 1]);
    __syncthreads();

    const int wave = tid >> 5, lane = tid & 31;
    const int n = lane & 15, sel = lane >> 4;          // half-wave select
    const uint4* ap = (const uint4*)sAh;
    const uint4* wp = (const uint4*)w16;

    for (int ti = wave; ti < 50; ti += 8) {
        const int mt = ti / 25, nt = ti % 25;
        const int g    = nt * 16 + n;                  // gate (column) index
        const int arow = mt * 16 + n;                  // panel row (M)
        __builtin_prefetch(wp + g * 96, 0, 1);         // global_prefetch_b8

        v8f acc = {};
        for (int kk = 0; kk < 24; ++kk) {
            // A frag: runs K = k0+sel*8+(0..7) and k0+16+sel*8+(0..7)
            AF a;
            const int ab = arow * 96 + kk * 4 + sel;
            a.q[0] = ap[ab];
            a.q[1] = ap[ab + 2];
            // B frag: K = k0+sel*8+(0..7) and k0+16+sel*8+(0..7)
            AF b;
            const int wb = g * 96 + kk * 4 + sel;
            b.q[0] = wp[wb];
            b.q[1] = wp[wb + 2];
            acc = __builtin_amdgcn_wmma_f32_16x16x32_f16(
                false, a.h, false, b.h, (short)0, acc, false, false);
        }
        const float bias  = b_ih[g] + b_hh[g];
        const long  rbase = row0 + mt * 16;
#pragma unroll
        for (int v = 0; v < 8; ++v) {                  // C/D: v -> M = v + sel*8
            const long r = rbase + v + sel * 8;
            xp[r * G4 + g] = acc[v] + bias;
        }
    }
}

// ---------------------------------------------------------------------------
// Kernel B: recurrent scan. 4 blocks x 256 threads; block owns 16 batch rows.
// ---------------------------------------------------------------------------
__global__ __launch_bounds__(256) void k_scan(
    const float* __restrict__ xp, const float* __restrict__ w_hh,
    float* __restrict__ hlast)
{
    __shared__ unsigned  sB[25 * 4 * 32 * 8];    // 100 KB weight fragments
    __shared__ _Float16  sH[16 * 128];           //   4 KB h (f16, K-padded)
    __shared__ float     sC[16 * HDIM];          // 6.4 KB cell state
    __shared__ float     sG[16 * G4];            //25.6 KB gates (async dst)
    __shared__ float     sHf[16 * HDIM];         // 6.4 KB h (f32)
    const int tid = threadIdx.x;
    const int bb  = blockIdx.x * 16;             // first batch row of block

    // Preload B fragments: frag(nt,kk), lane = N col (g), K = kk*32+sel*16+h.
    for (int idx = tid; idx < 25 * 4 * 32 * 8; idx += 256) {
        const int v     = idx & 7;
        const int lane_ = (idx >> 3) & 31;
        const int kk    = (idx >> 8) & 3;
        const int nt    = idx >> 10;
        const int g     = nt * 16 + (lane_ & 15);
        const int klo   = kk * 32 + ((lane_ >> 4) << 4) + 2 * v;
        const int khi   = klo + 1;
        const float lo  = (klo < HDIM) ? w_hh[g * HDIM + klo] : 0.0f;
        const float hi  = (khi < HDIM) ? w_hh[g * HDIM + khi] : 0.0f;
        sB[idx] = pack2h(lo, hi);
    }
    for (int i = tid; i < 16 * 128;  i += 256) sH[i] = (_Float16)0.0f;
    for (int i = tid; i < 16 * HDIM; i += 256) { sC[i] = 0.0f; sHf[i] = 0.0f; }
    __syncthreads();

    const int wave = tid >> 5, lane = tid & 31;
    const int n = lane & 15, sel = lane >> 4;
    const unsigned* sHu  = (const unsigned*)sH;
    const uint4*    sBq  = (const uint4*)sB;
    const unsigned  sGld = lds_off32(sG);

    for (int t = 0; t < T_LEN; ++t) {
        // ---- Phase X: gates = x_proj slice via async global->LDS ----
        for (int i = tid; i < 16 * 100; i += 256) {     // 1600 x 16B
            const int  b = i / 100, q = i % 100;
            const char* ga = (const char*)(xp +
                ((long)(bb + b) * T_LEN + t) * G4) + q * 16;
            async_b128(sGld + i * 16, ga);
        }
        wait_async0();
        __syncthreads();

        // ---- Phase R: gates += h @ w_hh^T via WMMA ----
        AF afr[4];
#pragma unroll
        for (int kk = 0; kk < 4; ++kk) {               // A frag: row = batch = n
            const int base = n * 64 + kk * 16 + sel * 8;
            afr[kk].q[0] = *(const uint4*)&sHu[base];
            afr[kk].q[1] = *(const uint4*)&sHu[base + 4];
        }
        for (int nt = wave; nt < 25; nt += 8) {
            v8f acc = {};
#pragma unroll
            for (int kk = 0; kk < 4; ++kk) {
                AF bf;
                const int fi = ((nt * 4 + kk) * 32 + lane) * 2;
                bf.q[0] = sBq[fi];
                bf.q[1] = sBq[fi + 1];
                acc = __builtin_amdgcn_wmma_f32_16x16x32_f16(
                    false, afr[kk].h, false, bf.h, (short)0, acc, false, false);
            }
            const int g = nt * 16 + n;
#pragma unroll
            for (int v = 0; v < 8; ++v) {
                const int M = v + sel * 8;             // batch row within block
                sG[M * G4 + g] += acc[v];
            }
        }
        __syncthreads();

        // ---- Phase N: LSTM cell update (gate order i,f,g,o) ----
        for (int idx = tid; idx < 16 * HDIM; idx += 256) {
            const int b = idx / HDIM, j = idx % HDIM;
            const float xi = sG[b * G4 + j];
            const float xf = sG[b * G4 + HDIM + j];
            const float xg = sG[b * G4 + 2 * HDIM + j];
            const float xo = sG[b * G4 + 3 * HDIM + j];
            const float ig = sigm(xi), fg = sigm(xf);
            const float gg = tanh_fast(xg), og = sigm(xo);
            const float c  = fg * sC[idx] + ig * gg;
            sC[idx] = c;
            const float h = og * tanh_fast(c);
            sHf[idx] = h;
            sH[b * 128 + j] = (_Float16)h;             // padded lanes stay zero
        }
        __syncthreads();
    }

    for (int idx = tid; idx < 16 * HDIM; idx += 256) {
        const int b = idx / HDIM, j = idx % HDIM;
        hlast[(bb + b) * HDIM + j] = sHf[idx];
    }
}

// ---------------------------------------------------------------------------
// Kernel C: tiny MLP head.
// ---------------------------------------------------------------------------
__global__ __launch_bounds__(256) void k_head(
    const float* __restrict__ hlast, const float* __restrict__ w1,
    const float* __restrict__ b1, const float* __restrict__ w2,
    const float* __restrict__ b2, float* __restrict__ out)
{
    __shared__ float sL[BATCH * HDIM];           // 25.6 KB
    const int tid = threadIdx.x;
    for (int idx = tid; idx < BATCH * HDIM; idx += 256) {
        const int b = idx / HDIM, j = idx % HDIM;
        float s = b1[j];
        const float* hr = hlast + b * HDIM;
        const float* wr = w1 + j * HDIM;
        for (int k = 0; k < HDIM; ++k) s += hr[k] * wr[k];
        sL[idx] = fmaxf(s, 0.0f);
    }
    __syncthreads();
    if (tid < BATCH) {
        float s = b2[0];
        const float* lr = sL + tid * HDIM;
        for (int j = 0; j < HDIM; ++j) s += lr[j] * w2[j];
        out[tid] = s;
    }
}

// ---------------------------------------------------------------------------
extern "C" void kernel_launch(void* const* d_in, const int* in_sizes, int n_in,
                              void* d_out, int out_size, void* d_ws, size_t ws_size,
                              hipStream_t stream) {
    const float* seq  = (const float*)d_in[0];
    const float* w_ih = (const float*)d_in[1];
    const float* w_hh = (const float*)d_in[2];
    const float* b_ih = (const float*)d_in[3];
    const float* b_hh = (const float*)d_in[4];
    const float* w1   = (const float*)d_in[5];
    const float* b1   = (const float*)d_in[6];
    const float* w2   = (const float*)d_in[7];
    const float* b2   = (const float*)d_in[8];
    float* out = (float*)d_out;

    float*    xp    = (float*)d_ws;                        // [B*T, 400] f32
    float*    hlast = xp + (size_t)BATCH * T_LEN * G4;     // [64, 100] f32
    unsigned* w16   = (unsigned*)(hlast + BATCH * HDIM);   // [400*768/2] f16x2

    k_prep<<<dim3((G4 * IN_DIM / 2 + 255) / 256), dim3(256), 0, stream>>>(
        w_ih, w16);
    k_xproj<<<dim3((BATCH * T_LEN) / 32), dim3(256), 0, stream>>>(
        seq, w16, b_ih, b_hh, xp);
    k_scan<<<dim3(4), dim3(256), 0, stream>>>(xp, w_hh, hlast);
    k_head<<<dim3(1), dim3(256), 0, stream>>>(hlast, w1, b1, w2, b2, out);
}
